// FeatureSimilarityAttention_30700426231853
// MI455X (gfx1250) — matrix operand order repaired and checked by reading, not compiled, not run on hardware
//
#include <hip/hip_runtime.h>
#include <hip/hip_bf16.h>
#include <math.h>

// ---------------------------------------------------------------------------
// FeatureSimilarityAttention for MI455X (gfx1250, wave32, WMMA f16->f32)
// B=8, L=512, D=512, N_FEAT=11
// GEMM: 8 waves/block (2x4), each wave owns a 32x64 C tile (2x4 WMMA frags)
// ---------------------------------------------------------------------------

typedef __attribute__((ext_vector_type(16))) _Float16 v16h;
typedef __attribute__((ext_vector_type(8)))  _Float16 v8h;
typedef __attribute__((ext_vector_type(8)))  float    v8f;

#define NFEAT 11
#define NEGC  (-1000000000.0f)

// ---------------------------------------------------------------------------
// elementwise f32 -> f16 conversion
// ---------------------------------------------------------------------------
__global__ __launch_bounds__(256) void k_f32_to_f16(const float* __restrict__ src,
                                                    _Float16* __restrict__ dst,
                                                    int n) {
  int i = blockIdx.x * 256 + threadIdx.x;
  if (i < n) dst[i] = (_Float16)src[i];
}

// ---------------------------------------------------------------------------
// feature-distance attention + row softmax, output f16 probabilities
// grid = (L, B), block = 256
// ---------------------------------------------------------------------------
__global__ __launch_bounds__(256) void k_feat_softmax(const float* __restrict__ x,
                                                      const float* __restrict__ fimp,
                                                      const int* __restrict__ lens,
                                                      _Float16* __restrict__ feat16,
                                                      int L, int D) {
  __shared__ float red[256];
  const int b = blockIdx.y;
  const int i = blockIdx.x;
  const int len = lens[b];
  const float* xb = x + ((size_t)b * L) * D;

  float fi[NFEAT], xi[NFEAT];
#pragma unroll
  for (int f = 0; f < NFEAT; ++f) {
    fi[f] = fimp[f];
    xi[f] = xb[(size_t)i * D + f];
  }

  float sc[2];
#pragma unroll
  for (int t = 0; t < 2; ++t) {
    int j = threadIdx.x + t * 256;
    const float* xj = xb + (size_t)j * D;
    float acc = 0.0f;
#pragma unroll
    for (int f = 0; f < NFEAT; ++f) acc += fabsf(xi[f] - xj[f]) * fi[f];
    sc[t] = (j < len) ? acc : NEGC;
  }

  // block max
  red[threadIdx.x] = fmaxf(sc[0], sc[1]);
  __syncthreads();
  for (int o = 128; o > 0; o >>= 1) {
    if (threadIdx.x < o) red[threadIdx.x] = fmaxf(red[threadIdx.x], red[threadIdx.x + o]);
    __syncthreads();
  }
  float mx = red[0];
  __syncthreads();

  float e0 = __expf(sc[0] - mx);
  float e1 = __expf(sc[1] - mx);

  red[threadIdx.x] = e0 + e1;
  __syncthreads();
  for (int o = 128; o > 0; o >>= 1) {
    if (threadIdx.x < o) red[threadIdx.x] += red[threadIdx.x + o];
    __syncthreads();
  }
  float inv = 1.0f / red[0];

  _Float16* fr = feat16 + ((size_t)b * L + i) * L;
  fr[threadIdx.x]        = (_Float16)(e0 * inv);
  fr[threadIdx.x + 256]  = (_Float16)(e1 * inv);
}

// ---------------------------------------------------------------------------
// WMMA fragment loads per CDNA5 ISA 7.12.2 (wave32, 16-bit operands)
// A 16x32: lane m=lane&15; K chunks {c0..c0+7, c0+16..c0+23}, c0=8*(lane>>4)
// B 32x16: lane n=lane&15; same K order as A (GEMM below loads A and B as 16 contiguous K at 16*(lane>>4))
// ---------------------------------------------------------------------------
__device__ __forceinline__ v16h load_frag_a(const _Float16* p) {
  v8h lo = *(const v8h*)(p);
  v8h hi = *(const v8h*)(p + 16);
  v16h r;
#pragma unroll
  for (int i = 0; i < 8; ++i) { r[i] = lo[i]; r[i + 8] = hi[i]; }
  return r;
}

// ---------------------------------------------------------------------------
// Generic batched WMMA GEMM:  C[m,n] = sum_k A[m,k] * Brow[n,k]
// Block = 256 threads = 8 waves arranged 2(M) x 4(N); block tile 64 x 256.
// Each wave: 32x64 C tile = 2x4 accumulators; per K-step 2 A-frags + 4 B-frags
// feed 8 WMMAs (0.75 fragment loads per WMMA).
// Epilogue modes:
//   0: v = acc*scaleC
//   1: v = acc*scaleC + Add[m,n]; v = (n < lens[b]) ? tanh(v) : 0   (attn)
//   2: v = acc*scaleC + Add[m,n]                                     (residual)
// Outputs (any may be null): Cf32 row-major, C16 row-major, C16T transposed.
// ---------------------------------------------------------------------------
__global__ __launch_bounds__(256) void k_wmma_gemm(
    const _Float16* __restrict__ A, const _Float16* __restrict__ Brow,
    size_t strideA, size_t strideB,
    int M, int N, int K, int lda, int ldb,
    float scaleC,
    float* __restrict__ Cf32, size_t strideCf, int ldcf,
    _Float16* __restrict__ C16, size_t strideC16, int ldc16,
    _Float16* __restrict__ C16T, size_t strideC16T, int ldc16T,
    int mode,
    const float* __restrict__ Add, size_t strideAdd, int ldadd,
    const int* __restrict__ lens) {
  const int lane  = threadIdx.x & 31;
  const int wave  = threadIdx.x >> 5;
  const int waveM = wave >> 2;                 // 0..1
  const int waveN = wave & 3;                  // 0..3
  const int tm0 = blockIdx.y * 64  + waveM * 32;
  const int tn0 = blockIdx.x * 256 + waveN * 64;
  const int b   = blockIdx.z;
  if (tm0 >= M || tn0 >= N) return;

  const _Float16* Ab = A + (size_t)b * strideA;
  const _Float16* Bb = Brow + (size_t)b * strideB;

  const int mr  = lane & 15;
  const int c0a = (lane >> 4) << 4;   // A chunk base: 0 or 16
  const int c0b = (lane >> 4) << 4;   // B chunk base: 0 or 16

  const _Float16* arow0 = Ab + (size_t)(tm0 + mr) * lda + c0a;
  const _Float16* arow1 = arow0 + (size_t)16 * lda;
  const _Float16* br0 = Bb + (size_t)(tn0 +  0 + mr) * ldb + c0b;
  const _Float16* br1 = br0 + (size_t)16 * ldb;
  const _Float16* br2 = br0 + (size_t)32 * ldb;
  const _Float16* br3 = br0 + (size_t)48 * ldb;

  v8f acc[2][4];
#pragma unroll
  for (int i = 0; i < 2; ++i)
#pragma unroll
    for (int j = 0; j < 4; ++j) acc[i][j] = (v8f){};

  for (int k0 = 0; k0 < K; k0 += 32) {
    // pull next cachelines of the streamed rows (global_prefetch_b8)
    __builtin_prefetch(arow0 + k0 + 256, 0, 1);
    __builtin_prefetch(br0   + k0 + 256, 0, 1);

    v16h a0 = *(const v16h*)(arow0 + k0);
    v16h a1 = *(const v16h*)(arow1 + k0);
    v16h b0 = *(const v16h*)(br0 + k0);
    v16h b1 = *(const v16h*)(br1 + k0);
    v16h b2 = *(const v16h*)(br2 + k0);
    v16h b3 = *(const v16h*)(br3 + k0);

    acc[0][0] = __builtin_amdgcn_wmma_f32_16x16x32_f16(false, a0, false, b0, (short)0, acc[0][0], false, false);
    acc[0][1] = __builtin_amdgcn_wmma_f32_16x16x32_f16(false, a0, false, b1, (short)0, acc[0][1], false, false);
    acc[0][2] = __builtin_amdgcn_wmma_f32_16x16x32_f16(false, a0, false, b2, (short)0, acc[0][2], false, false);
    acc[0][3] = __builtin_amdgcn_wmma_f32_16x16x32_f16(false, a0, false, b3, (short)0, acc[0][3], false, false);
    acc[1][0] = __builtin_amdgcn_wmma_f32_16x16x32_f16(false, a1, false, b0, (short)0, acc[1][0], false, false);
    acc[1][1] = __builtin_amdgcn_wmma_f32_16x16x32_f16(false, a1, false, b1, (short)0, acc[1][1], false, false);
    acc[1][2] = __builtin_amdgcn_wmma_f32_16x16x32_f16(false, a1, false, b2, (short)0, acc[1][2], false, false);
    acc[1][3] = __builtin_amdgcn_wmma_f32_16x16x32_f16(false, a1, false, b3, (short)0, acc[1][3], false, false);
  }

  const int len = (mode == 1) ? lens[b] : 0;

#pragma unroll
  for (int mi = 0; mi < 2; ++mi) {
    const int mbase = tm0 + mi * 16 + ((lane >> 4) << 3);
#pragma unroll
    for (int ni = 0; ni < 4; ++ni) {
      const int ncol = tn0 + ni * 16 + (lane & 15);
#pragma unroll
      for (int r = 0; r < 8; ++r) {
        const int mrow = mbase + r;
        float v = acc[mi][ni][r] * scaleC;
        if (mode == 1) {
          v += Add[(size_t)b * strideAdd + (size_t)mrow * ldadd + ncol];
          v = (ncol < len) ? tanhf(v) : 0.0f;
        } else if (mode == 2) {
          v += Add[(size_t)b * strideAdd + (size_t)mrow * ldadd + ncol];
        }
        if (Cf32) Cf32[(size_t)b * strideCf  + (size_t)mrow * ldcf  + ncol] = v;
        if (C16)  C16 [(size_t)b * strideC16 + (size_t)mrow * ldc16 + ncol] = (_Float16)v;
        if (C16T) C16T[(size_t)b * strideC16T + (size_t)ncol * ldc16T + mrow] = (_Float16)v;
      }
    }
  }
}

// ---------------------------------------------------------------------------
// LayerNorm over last dim (D=512). grid = B*L rows, block = 256.
// ---------------------------------------------------------------------------
__global__ __launch_bounds__(256) void k_layernorm(const float* __restrict__ in,
                                                   const float* __restrict__ g,
                                                   const float* __restrict__ bt,
                                                   float* __restrict__ out,
                                                   int D) {
  __shared__ float red[256];
  const int row = blockIdx.x;
  const float* r = in + (size_t)row * D;
  float v0 = r[threadIdx.x];
  float v1 = r[threadIdx.x + 256];

  red[threadIdx.x] = v0 + v1;
  __syncthreads();
  for (int o = 128; o > 0; o >>= 1) {
    if (threadIdx.x < o) red[threadIdx.x] += red[threadIdx.x + o];
    __syncthreads();
  }
  float mu = red[0] / (float)D;
  __syncthreads();

  float d0 = v0 - mu, d1 = v1 - mu;
  red[threadIdx.x] = d0 * d0 + d1 * d1;
  __syncthreads();
  for (int o = 128; o > 0; o >>= 1) {
    if (threadIdx.x < o) red[threadIdx.x] += red[threadIdx.x + o];
    __syncthreads();
  }
  float inv = rsqrtf(red[0] / (float)D + 1e-6f);

  out[(size_t)row * D + threadIdx.x]       = d0 * inv * g[threadIdx.x]       + bt[threadIdx.x];
  out[(size_t)row * D + threadIdx.x + 256] = d1 * inv * g[threadIdx.x + 256] + bt[threadIdx.x + 256];
}

// ---------------------------------------------------------------------------
// Host-side orchestration
// ---------------------------------------------------------------------------
extern "C" void kernel_launch(void* const* d_in, const int* in_sizes, int n_in,
                              void* d_out, int out_size, void* d_ws, size_t ws_size,
                              hipStream_t stream) {
  (void)in_sizes; (void)n_in; (void)out_size; (void)ws_size;
  const int B = 8, L = 512, D = 512;
  const size_t E  = (size_t)B * L * D;   // 2,097,152
  const size_t W  = (size_t)D * D;       //   262,144
  const size_t A2 = (size_t)B * L * L;   // 2,097,152
  const float invT = 1.0f / sqrtf((float)D);

  const float* q    = (const float*)d_in[0];
  const float* k    = (const float*)d_in[1];
  const float* v    = (const float*)d_in[2];
  const float* x    = (const float*)d_in[3];
  const int*   lens = (const int*)  d_in[4];
  const float* Wq   = (const float*)d_in[5];
  const float* Wk   = (const float*)d_in[6];
  const float* Wv   = (const float*)d_in[7];
  const float* Wqf  = (const float*)d_in[8];
  const float* Wkf  = (const float*)d_in[9];
  const float* Wfc  = (const float*)d_in[10];
  const float* fimp = (const float*)d_in[11];
  const float* gam  = (const float*)d_in[12];
  const float* bet  = (const float*)d_in[13];

  float* out_norm = (float*)d_out;        // [B,L,D]
  float* out_attn = out_norm + E;         // [B,L,L]

  // ---- workspace carve-up ----
  char* p = (char*)d_ws;
  auto give = [&](size_t bytes) -> char* {
    char* r = p;
    p += (bytes + 255) & ~(size_t)255;
    return r;
  };
  _Float16* qh    = (_Float16*)give(E * 2);
  _Float16* kh    = (_Float16*)give(E * 2);
  _Float16* vh    = (_Float16*)give(E * 2);
  _Float16* Wq16  = (_Float16*)give(W * 2);
  _Float16* Wk16  = (_Float16*)give(W * 2);
  _Float16* Wv16  = (_Float16*)give(W * 2);
  _Float16* Wqf16 = (_Float16*)give(W * 2);
  _Float16* Wkf16 = (_Float16*)give(W * 2);
  _Float16* Wfc16 = (_Float16*)give(W * 2);
  _Float16* qp16  = (_Float16*)give(E * 2);   // (q@Wq^T)/sqrt(D), row-major
  _Float16* kp16  = (_Float16*)give(E * 2);   // k@Wk^T, row-major
  _Float16* vpT   = (_Float16*)give(E * 2);   // (v@Wv^T)^T  [B,D,L]
  _Float16* qfT   = (_Float16*)give(E * 2);   // (q@Wqf^T)^T [B,D,L]
  _Float16* kfT   = (_Float16*)give(E * 2);   // (k@Wkf^T)^T [B,D,L]
  _Float16* feat16  = (_Float16*)give(A2 * 2); // softmax(feat_attn) f16
  _Float16* qf2_16  = (_Float16*)give(E * 2);  // (feat@qf)/sqrt(D)
  _Float16* kf2_16  = (_Float16*)give(E * 2);
  _Float16* attn16  = (_Float16*)give(A2 * 2); // tanh-masked attention f16
  _Float16* out1_16 = (_Float16*)give(E * 2);  // attn@vp
  float*    attnf   = (float*)give(A2 * 4);    // qp@kp^T accumulator
  float*    outf    = (float*)give(E * 4);     // pre-LN output

  // ---- 1. f32 -> f16 conversions ----
  {
    dim3 blk(256);
    k_f32_to_f16<<<dim3((unsigned)(E / 256)), blk, 0, stream>>>(q, qh, (int)E);
    k_f32_to_f16<<<dim3((unsigned)(E / 256)), blk, 0, stream>>>(k, kh, (int)E);
    k_f32_to_f16<<<dim3((unsigned)(E / 256)), blk, 0, stream>>>(v, vh, (int)E);
    k_f32_to_f16<<<dim3((unsigned)(W / 256)), blk, 0, stream>>>(Wq,  Wq16,  (int)W);
    k_f32_to_f16<<<dim3((unsigned)(W / 256)), blk, 0, stream>>>(Wk,  Wk16,  (int)W);
    k_f32_to_f16<<<dim3((unsigned)(W / 256)), blk, 0, stream>>>(Wv,  Wv16,  (int)W);
    k_f32_to_f16<<<dim3((unsigned)(W / 256)), blk, 0, stream>>>(Wqf, Wqf16, (int)W);
    k_f32_to_f16<<<dim3((unsigned)(W / 256)), blk, 0, stream>>>(Wkf, Wkf16, (int)W);
    k_f32_to_f16<<<dim3((unsigned)(W / 256)), blk, 0, stream>>>(Wfc, Wfc16, (int)W);
  }

  const dim3 gblk(256);
  const dim3 ggrid(2, 8, 8);   // N/256, M/64, B  (all GEMMs are 512^3)
  const size_t sLD = (size_t)L * D, sLL = (size_t)L * L, sDL = (size_t)D * L;

  // ---- 2. feature softmax (f32 math, f16 probabilities) ----
  k_feat_softmax<<<dim3(L, B), gblk, 0, stream>>>(x, fimp, lens, feat16, L, D);

  // ---- 3. projections: C = X @ W^T (B-operand column n == W row n) ----
  // qp (scaled 1/sqrt(D)), kp : f16 row-major
  k_wmma_gemm<<<ggrid, gblk, 0, stream>>>(qh, Wq16, sLD, 0, L, D, D, D, D, invT,
      nullptr, 0, 0, qp16, sLD, D, nullptr, 0, 0, 0, nullptr, 0, 0, nullptr);
  k_wmma_gemm<<<ggrid, gblk, 0, stream>>>(kh, Wk16, sLD, 0, L, D, D, D, D, 1.0f,
      nullptr, 0, 0, kp16, sLD, D, nullptr, 0, 0, 0, nullptr, 0, 0, nullptr);
  // vp, qf, kf : f16 transposed [B,D,L] so they can serve as B-operands later
  k_wmma_gemm<<<ggrid, gblk, 0, stream>>>(vh, Wv16, sLD, 0, L, D, D, D, D, 1.0f,
      nullptr, 0, 0, nullptr, 0, 0, vpT, sDL, L, 0, nullptr, 0, 0, nullptr);
  k_wmma_gemm<<<ggrid, gblk, 0, stream>>>(qh, Wqf16, sLD, 0, L, D, D, D, D, 1.0f,
      nullptr, 0, 0, nullptr, 0, 0, qfT, sDL, L, 0, nullptr, 0, 0, nullptr);
  k_wmma_gemm<<<ggrid, gblk, 0, stream>>>(kh, Wkf16, sLD, 0, L, D, D, D, D, 1.0f,
      nullptr, 0, 0, nullptr, 0, 0, kfT, sDL, L, 0, nullptr, 0, 0, nullptr);

  // ---- 4. attn1 = (qp/sqrt(D)) @ kp^T  -> f32 accumulator ----
  k_wmma_gemm<<<ggrid, gblk, 0, stream>>>(qp16, kp16, sLD, sLD, L, L, D, D, D, 1.0f,
      attnf, sLL, L, nullptr, 0, 0, nullptr, 0, 0, 0, nullptr, 0, 0, nullptr);

  // ---- 5. qf2 = feat @ qf (scaled 1/sqrt(D)),  kf2 = feat @ kf ----
  k_wmma_gemm<<<ggrid, gblk, 0, stream>>>(feat16, qfT, sLL, sDL, L, D, L, L, L, invT,
      nullptr, 0, 0, qf2_16, sLD, D, nullptr, 0, 0, 0, nullptr, 0, 0, nullptr);
  k_wmma_gemm<<<ggrid, gblk, 0, stream>>>(feat16, kfT, sLL, sDL, L, D, L, L, L, 1.0f,
      nullptr, 0, 0, kf2_16, sLD, D, nullptr, 0, 0, 0, nullptr, 0, 0, nullptr);

  // ---- 6. attn = tanh(mask(attn1 + qf2 @ kf2^T)) -> d_out(attn) f32 + f16 ----
  k_wmma_gemm<<<ggrid, gblk, 0, stream>>>(qf2_16, kf2_16, sLD, sLD, L, L, D, D, D, 1.0f,
      out_attn, sLL, L, attn16, sLL, L, nullptr, 0, 0, 1, attnf, sLL, L, lens);

  // ---- 7. out1 = attn @ vp  (B-operand = vpT rows) -> f16 ----
  k_wmma_gemm<<<ggrid, gblk, 0, stream>>>(attn16, vpT, sLL, sDL, L, D, L, L, L, 1.0f,
      nullptr, 0, 0, out1_16, sLD, D, nullptr, 0, 0, 0, nullptr, 0, 0, nullptr);

  // ---- 8. out = out1 @ Wfc^T + residual(q) -> f32 ----
  k_wmma_gemm<<<ggrid, gblk, 0, stream>>>(out1_16, Wfc16, sLD, 0, L, D, D, D, D, 1.0f,
      outf, sLD, D, nullptr, 0, 0, nullptr, 0, 0, 2, q, sLD, D, nullptr);

  // ---- 9. LayerNorm -> d_out(out) ----
  k_layernorm<<<dim3(B * L), gblk, 0, stream>>>(outf, gam, bet, out_norm, D);
}
